// GroupedLossWithIndexMap_89300960018699
// MI455X (gfx1250) — compile-verified, matrix operand order checked by reading, not run on hardware
//
#include <hip/hip_runtime.h>
#include <hip/hip_bf16.h>

typedef __attribute__((ext_vector_type(16))) _Float16 v16h;
typedef __attribute__((ext_vector_type(8)))  float    v8f;

#define C_OLD   128
#define C_NEW   16
#define B_SZ    8
#define N_ITEMS 65536

// ---------------------------------------------------------------------------
// Kernel 0: zero the [B][C_NEW] f32 accumulator in workspace (ws is poisoned).
// ---------------------------------------------------------------------------
__global__ void gl_init_ws(float* __restrict__ acc) {
    acc[threadIdx.x] = 0.0f;
}

// ---------------------------------------------------------------------------
// Kernel 1: streaming pass. One wave owns a 16-item tile.
//   A (16x32 f16, per CDNA5 A-layout) = exp(logits) for a 32-class K-chunk
//   B (32x16 f16, constant)           = one-hot group matrix chunk
//   D = A*B + C (f32)                 = grouped exp-sums [16 items x 16 groups]
// 4 chained WMMAs cover K=128. Per-item totals via 16-lane shfl_xor over D,
// normalized grouped probs accumulated in f32 registers across the tile loop,
// then one atomicAdd per (batch, group) per wave.
// ---------------------------------------------------------------------------
__global__ __launch_bounds__(256) void gl_grouped_softmax_sum(
        const float* __restrict__ inputs, float* __restrict__ acc) {
    const int lane  = threadIdx.x & 31;
    const int wv    = threadIdx.x >> 5;
    const int batch = blockIdx.y;
    const int n     = lane & 15;              // item row (A) / group column (B,D)
    const int klo   = (lane < 16) ? 0 : 8;    // K sub-block per ISA 16-bit A layout

    // Constant one-hot group matrices for the 4 K-chunks (B-layout mirrors A).
    v16h Bm[4];
    #pragma unroll
    for (int c = 0; c < 4; ++c) {
        #pragma unroll
        for (int j = 0; j < 16; ++j) {
            // half j -> K = klo + j (j<8) or klo + 16 + (j-8)
            const int k   = klo + ((j < 8) ? j : (8 + j));
            const int cls = c * 32 + k;
            Bm[c][j] = (n == (cls >> 3)) ? (_Float16)1.0f : (_Float16)0.0f;
        }
    }

    float accr[8];
    #pragma unroll
    for (int r = 0; r < 8; ++r) accr[r] = 0.0f;

    const int tiles       = N_ITEMS / 16;
    const int waves       = gridDim.x * (blockDim.x >> 5);
    const int gw          = blockIdx.x * (blockDim.x >> 5) + wv;
    const float* bbase    = inputs + (size_t)batch * N_ITEMS * C_OLD;

    for (int t = gw; t < tiles; t += waves) {
        const float* p = bbase + (size_t)(t * 16 + n) * C_OLD;
        v8f d = {};                           // f32 accumulator (C starts at 0)
        #pragma unroll
        for (int c = 0; c < 4; ++c) {
            const float4* p0 = (const float4*)(p + c * 32 + klo);
            const float4* p1 = (const float4*)(p + c * 32 + klo + 16);
            const float4 x0 = p0[0], x1 = p0[1];
            const float4 y0 = p1[0], y1 = p1[1];
            v16h a;
            a[0]  = (_Float16)__expf(x0.x); a[1]  = (_Float16)__expf(x0.y);
            a[2]  = (_Float16)__expf(x0.z); a[3]  = (_Float16)__expf(x0.w);
            a[4]  = (_Float16)__expf(x1.x); a[5]  = (_Float16)__expf(x1.y);
            a[6]  = (_Float16)__expf(x1.z); a[7]  = (_Float16)__expf(x1.w);
            a[8]  = (_Float16)__expf(y0.x); a[9]  = (_Float16)__expf(y0.y);
            a[10] = (_Float16)__expf(y0.z); a[11] = (_Float16)__expf(y0.w);
            a[12] = (_Float16)__expf(y1.x); a[13] = (_Float16)__expf(y1.y);
            a[14] = (_Float16)__expf(y1.z); a[15] = (_Float16)__expf(y1.w);
            d = __builtin_amdgcn_wmma_f32_16x16x32_f16(
                    false, a, false, Bm[c], (short)0, d, false, false);
        }
        // Per-item totals: sum D over the 16 group columns (one 16-lane half
        // holds items 0..7 in VGPRs 0..7, the other half items 8..15).
        #pragma unroll
        for (int r = 0; r < 8; ++r) {
            float tt = d[r];
            tt += __shfl_xor(tt, 1, 16);
            tt += __shfl_xor(tt, 2, 16);
            tt += __shfl_xor(tt, 4, 16);
            tt += __shfl_xor(tt, 8, 16);
            accr[r] += d[r] / tt;             // normalized grouped probability
        }
    }

    float s = 0.0f;
    #pragma unroll
    for (int r = 0; r < 8; ++r) s += accr[r];
    s += __shfl_xor(s, 16, 32);               // merge item rows 0-7 and 8-15
    if (lane < 16) atomicAdd(&acc[batch * C_NEW + lane], s);
}

// ---------------------------------------------------------------------------
// Kernel 2: epilogue on [B=8, C_NEW=16]. thread t -> (b = t/16, g = t%16);
// 16-wide shfl reductions stay inside a wave32 half.
// ---------------------------------------------------------------------------
__global__ void gl_finalize(const float* __restrict__ acc,
                            const float* __restrict__ targets,
                            float* __restrict__ out) {
    __shared__ float kl_s[B_SZ];
    const int t = threadIdx.x;                // 0..127
    const int b = t >> 4, g = t & 15;

    const float avg = acc[t] / (float)N_ITEMS;

    // softmax(avg) over 16 groups
    float m = avg;
    m = fmaxf(m, __shfl_xor(m, 1, 16));
    m = fmaxf(m, __shfl_xor(m, 2, 16));
    m = fmaxf(m, __shfl_xor(m, 4, 16));
    m = fmaxf(m, __shfl_xor(m, 8, 16));
    float e = expf(avg - m);
    float s = e;
    s += __shfl_xor(s, 1, 16);
    s += __shfl_xor(s, 2, 16);
    s += __shfl_xor(s, 4, 16);
    s += __shfl_xor(s, 8, 16);
    const float p = e / s;

    // softmax(targets/100) over 16 groups
    const float tv = targets[t] * 0.01f;
    float mt = tv;
    mt = fmaxf(mt, __shfl_xor(mt, 1, 16));
    mt = fmaxf(mt, __shfl_xor(mt, 2, 16));
    mt = fmaxf(mt, __shfl_xor(mt, 4, 16));
    mt = fmaxf(mt, __shfl_xor(mt, 8, 16));
    float et = expf(tv - mt);
    float st = et;
    st += __shfl_xor(st, 1, 16);
    st += __shfl_xor(st, 2, 16);
    st += __shfl_xor(st, 4, 16);
    st += __shfl_xor(st, 8, 16);
    const float q = et / st;

    // KL term; sum over 16 groups, divide by C_NEW (batchmean over 1-D input)
    float kl = q * (logf(q) - logf(p + 1e-8f));
    kl += __shfl_xor(kl, 1, 16);
    kl += __shfl_xor(kl, 2, 16);
    kl += __shfl_xor(kl, 4, 16);
    kl += __shfl_xor(kl, 8, 16);
    if (g == 0) kl_s[b] = kl / (float)C_NEW;
    __syncthreads();
    if (t == 0) {
        float r = 0.0f;
        #pragma unroll
        for (int i = 0; i < B_SZ; ++i) r += kl_s[i];
        out[0] = r / (float)B_SZ;
    }
}

// ---------------------------------------------------------------------------
extern "C" void kernel_launch(void* const* d_in, const int* in_sizes, int n_in,
                              void* d_out, int out_size, void* d_ws, size_t ws_size,
                              hipStream_t stream) {
    (void)in_sizes; (void)n_in; (void)out_size; (void)ws_size;
    const float* inputs  = (const float*)d_in[0];   // [8, 65536, 128] f32
    const float* targets = (const float*)d_in[1];   // [8, 16] f32
    float* out = (float*)d_out;                     // scalar f32
    float* acc = (float*)d_ws;                      // [8*16] f32 accumulator

    gl_init_ws<<<1, B_SZ * C_NEW, 0, stream>>>(acc);
    dim3 grid(64, B_SZ);                            // 512 waves/batch-dim total
    gl_grouped_softmax_sum<<<grid, 256, 0, stream>>>(inputs, acc);
    gl_finalize<<<1, 128, 0, stream>>>(acc, targets, out);
}
